// Loss_reference_neg_2_85942295593596
// MI455X (gfx1250) — matrix operand order repaired and checked by reading, not compiled, not run on hardware
//
#include <hip/hip_runtime.h>
#include <hip/hip_bf16.h>

typedef __attribute__((ext_vector_type(16))) __bf16 v16bf;
typedef __attribute__((ext_vector_type(8)))  float  v8f;

#define NTHREADS 256

// -------- problem-size constants (from the reference) --------
#define BB    64
#define LL    256
#define DD    512
#define NCLS  10
#define NREF  50
#define NNEG  100
#define MARGIN 10.0f
#define EPS_FINAL 0.0025f
#define NSCHED 27   // 12 halvings (8.0 .. 0.00390625) + 15 @ 0.0025

__device__ inline int selidx(int mode, int p, const int* __restrict__ idx, int b) {
  if (mode == 0) return b / p;      // block / p
  if (mode == 1) return b % p;      // block % p
  if (mode == 2) return p;          // fixed index
  return idx[b];                    // indirect
}

// Load a 16-element bf16 fragment (hi + residual lo) from 16 f32 values.
// gap==0: 16 contiguous floats (unused: A and B both use gap==1)
// gap==1: floats [0..7] and [16..23] (A fragment: K = kbase..+7, kbase+16..+23)
// Rows are pre-clamped by the caller, so no guard/zero path is needed here.
__device__ inline void load_frag(const float* __restrict__ base, int gap,
                                 v16bf& hi, v16bf& lo) {
  const float4* p = (const float4*)base;
  float4 q0 = p[0], q1 = p[1];
  float4 q2 = gap ? p[4] : p[2];
  float4 q3 = gap ? p[5] : p[3];
  float v[16] = { q0.x,q0.y,q0.z,q0.w, q1.x,q1.y,q1.z,q1.w,
                  q2.x,q2.y,q2.z,q2.w, q3.x,q3.y,q3.z,q3.w };
#pragma unroll
  for (int j = 0; j < 16; ++j) {
    float x = v[j];
    __bf16 h = (__bf16)x;
    hi[j] = h;
    lo[j] = (__bf16)(x - (float)h);
  }
}

// One workgroup = one entropic-OT problem: cost matrix via bf16x3 WMMA into LDS,
// then 27 log-domain Sinkhorn iterations, then <a,f>+<b,g>.
__global__ __launch_bounds__(NTHREADS) void ot_kernel(
    const float* __restrict__ xbase, long long xstride,
    const float* __restrict__ ybase, long long ystride,
    int xm, int xp, int ym, int yp,
    const int* __restrict__ idx,
    const float* __restrict__ wa,   // x-cloud weights (row stride N) or null => uniform 1/N
    const float* __restrict__ wb,   // same for y (stride M) or null => uniform 1/M
    int N, int M, int D,
    float* __restrict__ out)
{
  extern __shared__ float smem[];
  const int tid = threadIdx.x, nth = blockDim.x;
  const int b = blockIdx.x;
  const int Np = (N + 15) & ~15;
  const int Mp = (M + 15) & ~15;
  const int Mpad = Mp + 1;            // odd stride -> conflict-free row access

  float* Csh = smem;                  // Np * Mpad
  float* f   = Csh + Np * Mpad;       // Np
  float* g   = f + Np;                // Mp
  float* la  = g + Mp;                // Np
  float* lb  = la + Np;               // Mp
  float* xx  = lb + Mp;               // Np
  float* yy  = xx + Np;               // Mp
  float* ha  = yy + Mp;               // Np : la[m] + f[m]/eps scratch
  float* hb  = ha + Np;               // Mp : lb[n] + g[n]/eps scratch
  float* red = hb + Mp;               // nth

  const int xs = selidx(xm, xp, idx, b);
  const int ys = selidx(ym, yp, idx, b);
  const float* X  = xbase + (long long)xs * xstride;
  const float* Y  = ybase + (long long)ys * ystride;
  const float* Aw = wa ? (wa + (long long)xs * N) : nullptr;
  const float* Bw = wb ? (wb + (long long)ys * M) : nullptr;

  // ---- squared norms, log-weights, potentials init ----
  for (int m = tid; m < Np; m += nth) {
    float s = 0.0f;
    if (m < N) {
      const float4* p = (const float4*)(X + (long long)m * D);
      for (int k = 0; k < (D >> 2); ++k) { float4 v = p[k]; s += v.x*v.x + v.y*v.y + v.z*v.z + v.w*v.w; }
    }
    xx[m] = s;
    float av = (m < N) ? (Aw ? Aw[m] : (1.0f / (float)N)) : 0.0f;
    la[m] = (av > 0.0f) ? __logf(fmaxf(av, 1e-30f)) : -1e9f;
    f[m] = 0.0f;
  }
  for (int n = tid; n < Mp; n += nth) {
    float s = 0.0f;
    if (n < M) {
      const float4* p = (const float4*)(Y + (long long)n * D);
      for (int k = 0; k < (D >> 2); ++k) { float4 v = p[k]; s += v.x*v.x + v.y*v.y + v.z*v.z + v.w*v.w; }
    }
    yy[n] = s;
    float bv = (n < M) ? (Bw ? Bw[n] : (1.0f / (float)M)) : 0.0f;
    lb[n] = (bv > 0.0f) ? __logf(fmaxf(bv, 1e-30f)) : -1e9f;
    g[n] = 0.0f;
  }
  __syncthreads();

  // ---- cost matrix: C[m][n] = 0.5*(xx[m]+yy[n]) - x_m . y_n  (bf16x3 WMMA) ----
  // Out-of-range rows are clamped (their products land only in pad cells of Csh
  // that the Sinkhorn loops never read), keeping the K-loop branch-free.
  {
    const int wave = tid >> 5, nW = nth >> 5, lane = tid & 31;
    const int half = lane >> 4, l16 = lane & 15;
    const int tilesN = Np >> 4, tilesM = Mp >> 4;
    const int kbaseA = half * 8;      // A: K = kbaseA..+7 and kbaseA+16..+23
    const int khalfB = half * 8;      // B: K = khalfB..+7 and khalfB+16..+23
    for (int t = wave; t < tilesN * tilesM; t += nW) {
      const int tm = t / tilesM, tn = t % tilesM;
      const int arow = min(tm * 16 + l16, N - 1);   // x row for A fragment (clamped)
      const int brow = min(tn * 16 + l16, M - 1);   // y row for B fragment (clamped)
      const float* Xp = X + (long long)arow * D + kbaseA;
      const float* Yp = Y + (long long)brow * D + khalfB;
      v8f acc = {};
      for (int k0 = 0; k0 < D; k0 += 32) {
        v16bf ahi, alo, bhi, blo;
        load_frag(Xp + k0, 1, ahi, alo);
        load_frag(Yp + k0, 1, bhi, blo);
        acc = __builtin_amdgcn_wmma_f32_16x16x32_bf16(false, ahi, false, bhi, (short)0, acc, false, false);
        acc = __builtin_amdgcn_wmma_f32_16x16x32_bf16(false, ahi, false, blo, (short)0, acc, false, false);
        acc = __builtin_amdgcn_wmma_f32_16x16x32_bf16(false, alo, false, bhi, (short)0, acc, false, false);
      }
      // f32 D layout: VGPR r -> M = r + 8*half, N = lane&15
      const int nc = tn * 16 + l16;
#pragma unroll
      for (int r = 0; r < 8; ++r) {
        const int mr = tm * 16 + r + 8 * half;
        Csh[mr * Mpad + nc] = 0.5f * (xx[mr] + yy[nc]) - acc[r];
      }
    }
  }
  __syncthreads();

  // ---- log-domain Sinkhorn with eps-scaling ----
  for (int it = 0; it < NSCHED; ++it) {
    const float eps = (it < 12) ? (8.0f / (float)(1 << it)) : EPS_FINAL;
    const float inv = 1.0f / eps;

    // hb[n] = lb[n] + g[n]/eps (hoisted out of the per-row inner loops)
    for (int n = tid; n < M; n += nth) hb[n] = lb[n] + g[n] * inv;
    __syncthreads();
    // f_m = -eps * lse_n( hb[n] - C_mn/eps )
    for (int m = tid; m < N; m += nth) {
      const float* Cr = Csh + m * Mpad;
      float vmax = -3.4e38f;
      for (int n = 0; n < M; ++n) vmax = fmaxf(vmax, hb[n] - Cr[n] * inv);
      float s = 0.0f;
      for (int n = 0; n < M; ++n) s += __expf(hb[n] - Cr[n] * inv - vmax);
      f[m] = -eps * (vmax + __logf(s));
    }
    __syncthreads();

    // ha[m] = la[m] + f[m]/eps (uses new f)
    for (int m = tid; m < N; m += nth) ha[m] = la[m] + f[m] * inv;
    __syncthreads();
    // g_n = -eps * lse_m( ha[m] - C_mn/eps )
    for (int n = tid; n < M; n += nth) {
      float vmax = -3.4e38f;
      for (int m = 0; m < N; ++m) vmax = fmaxf(vmax, ha[m] - Csh[m * Mpad + n] * inv);
      float s = 0.0f;
      for (int m = 0; m < N; ++m) s += __expf(ha[m] - Csh[m * Mpad + n] * inv - vmax);
      g[n] = -eps * (vmax + __logf(s));
    }
    __syncthreads();
  }

  // ---- OT value <a,f> + <b,g> ----
  float part = 0.0f;
  for (int m = tid; m < N; m += nth) part += (Aw ? Aw[m] : (1.0f / (float)N)) * f[m];
  for (int n = tid; n < M; n += nth) part += (Bw ? Bw[n] : (1.0f / (float)M)) * g[n];
  red[tid] = part;
  __syncthreads();
  for (int s = nth >> 1; s > 0; s >>= 1) {
    if (tid < s) red[tid] += red[tid + s];
    __syncthreads();
  }
  if (tid == 0) out[b] = red[0];
}

// Debiased scores, hinges, mean.
__global__ __launch_bounds__(64) void combine_kernel(
    const float* __restrict__ cross, const float* __restrict__ aa,
    const float* __restrict__ tt,    const float* __restrict__ nn,
    const float* __restrict__ ln,    const float* __restrict__ gnv,
    const int* __restrict__ grade,   const int* __restrict__ grade_neg,
    float* __restrict__ out)
{
  __shared__ float red[64];
  const int i = threadIdx.x;
  float Srow[NCLS];
  const float aai = aa[i];
#pragma unroll
  for (int c = 0; c < NCLS; ++c) Srow[c] = cross[i * NCLS + c] - 0.5f * aai - 0.5f * tt[c];
  const int gi = grade[i];
  const float pos = Srow[gi];
  float hinge = 0.0f;
#pragma unroll
  for (int c = 0; c < NCLS; ++c)
    if (c != gi) hinge += fmaxf(pos - Srow[c] + MARGIN, 0.0f);
  const float Slast = ln[i]  - 0.5f * tt[NCLS - 1]      - 0.5f * nn[i];
  const float Sgn   = gnv[i] - 0.5f * tt[grade_neg[i]]  - 0.5f * nn[i];
  red[i] = hinge + fmaxf(pos - Slast + MARGIN, 0.0f) + fabsf(Sgn - Slast);
  __syncthreads();
  for (int s = 32; s > 0; s >>= 1) {
    if (i < s) red[i] += red[i + s];
    __syncthreads();
  }
  if (i == 0) out[0] = red[0] / (float)BB;
}

static inline size_t ot_lds_bytes(int N, int M) {
  int Np = (N + 15) & ~15, Mp = (M + 15) & ~15, Mpad = Mp + 1;
  return (size_t)(Np * Mpad + 4 * Np + 4 * Mp + NTHREADS) * sizeof(float);
}

extern "C" void kernel_launch(void* const* d_in, const int* in_sizes, int n_in,
                              void* d_out, int out_size, void* d_ws, size_t ws_size,
                              hipStream_t stream) {
  const float* anchor  = (const float*)d_in[0];   // [B, L, D]
  const int*   grade   = (const int*)  d_in[2];   // [B]
  const float* weight  = (const float*)d_in[3];   // [B, L]
  const float* negs    = (const float*)d_in[4];   // [B, NNEG, D]
  const int*   gneg    = (const int*)  d_in[5];   // [B]
  const float* t0      = (const float*)d_in[6];   // [C, NREF, D]
  float* out = (float*)d_out;

  float* ws = (float*)d_ws;
  float* ws_cross = ws;                 // B*C
  float* ws_aa    = ws_cross + BB * NCLS;
  float* ws_tt    = ws_aa + BB;
  float* ws_nn    = ws_tt + NCLS;
  float* ws_ln    = ws_nn + BB;
  float* ws_gn    = ws_ln + BB;

  // Allow >64KB dynamic LDS (largest case: aa, 256x256 -> ~271KB; WGP has 320KB).
  const int max_lds = (int)ot_lds_bytes(LL, LL);
  (void)hipFuncSetAttribute(reinterpret_cast<const void*>(ot_kernel),
                            hipFuncAttributeMaxDynamicSharedMemorySize, max_lds);

  const long long sA = (long long)LL * DD;     // anchor row-cloud stride
  const long long sT = (long long)NREF * DD;   // t0 cloud stride
  const long long sN = (long long)NNEG * DD;   // neg cloud stride

  // ot_cross[i][c] = OT(weight_i, anchor_i ; 1/NREF, t0_c)
  ot_kernel<<<BB * NCLS, NTHREADS, ot_lds_bytes(LL, NREF), stream>>>(
      anchor, sA, t0, sT,
      /*xm*/0, /*xp*/NCLS, /*ym*/1, /*yp*/NCLS, nullptr,
      weight, nullptr, LL, NREF, DD, ws_cross);

  // ot_aa[i] = OT(weight_i, anchor_i ; weight_i, anchor_i)
  ot_kernel<<<BB, NTHREADS, ot_lds_bytes(LL, LL), stream>>>(
      anchor, sA, anchor, sA,
      0, 1, 0, 1, nullptr,
      weight, weight, LL, LL, DD, ws_aa);

  // ot_tt[c] = OT(u, t0_c ; u, t0_c)
  ot_kernel<<<NCLS, NTHREADS, ot_lds_bytes(NREF, NREF), stream>>>(
      t0, sT, t0, sT,
      0, 1, 0, 1, nullptr,
      nullptr, nullptr, NREF, NREF, DD, ws_tt);

  // ot_nn[i] = OT(u, neg_i ; u, neg_i)
  ot_kernel<<<BB, NTHREADS, ot_lds_bytes(NNEG, NNEG), stream>>>(
      negs, sN, negs, sN,
      0, 1, 0, 1, nullptr,
      nullptr, nullptr, NNEG, NNEG, DD, ws_nn);

  // ot_ln[i] = OT(u, t0[C-1] ; u, neg_i)
  ot_kernel<<<BB, NTHREADS, ot_lds_bytes(NREF, NNEG), stream>>>(
      t0, sT, negs, sN,
      /*xm*/2, /*xp*/NCLS - 1, 0, 1, nullptr,
      nullptr, nullptr, NREF, NNEG, DD, ws_ln);

  // ot_gn[i] = OT(u, t0[grade_neg[i]] ; u, neg_i)
  ot_kernel<<<BB, NTHREADS, ot_lds_bytes(NREF, NNEG), stream>>>(
      t0, sT, negs, sN,
      /*xm*/3, /*xp*/0, 0, 1, gneg,
      nullptr, nullptr, NREF, NNEG, DD, ws_gn);

  combine_kernel<<<1, 64, 0, stream>>>(ws_cross, ws_aa, ws_tt, ws_nn, ws_ln, ws_gn,
                                       grade, gneg, out);
}
